// MultiHeadAttention_10728828305869
// MI455X (gfx1250) — compile-verified, matrix operand order checked by reading, not run on hardware
//
#include <hip/hip_runtime.h>

// ---------------- problem constants (match reference) ----------------
constexpr int HEADS    = 8;
constexpr int HIDDEN   = 256;
constexpr int HEAD_DIM = 32;   // HIDDEN / HEADS
constexpr int BATCH    = 8;
constexpr int QNUM     = 512;
constexpr int VNUM     = 4096;

typedef __attribute__((ext_vector_type(8)))  float        v8f;
typedef __attribute__((ext_vector_type(16))) __bf16       v16bf;
typedef __attribute__((ext_vector_type(4)))  unsigned int u32x4;
typedef __attribute__((ext_vector_type(2)))  unsigned int u32x2;
typedef __attribute__((ext_vector_type(4)))  float        f32x4;

struct B32x8 { u32x4 lo, hi; };   // 32 bytes == one v16bf operand

// float -> bf16 (round-to-nearest-even)
__device__ __forceinline__ unsigned short f2bfbits(float f) {
  union { float f; unsigned u; } v; v.f = f;
  unsigned r = v.u + 0x7FFFu + ((v.u >> 16) & 1u);
  return (unsigned short)(r >> 16);
}
__device__ __forceinline__ __bf16 f2bf(float f) {
  unsigned short h = f2bfbits(f);
  return __builtin_bit_cast(__bf16, h);
}

// CDNA5 16-bit A/B operand layout (16xK tile, wave32):
//   lanes 0-15  : M/N = lane,    elems 0..7 -> K = 0..7,   8..15 -> K = 16..23
//   lanes 16-31 : M/N = lane-16, elems 0..7 -> K = 8..15,  8..15 -> K = 24..31
// With bf16 stored K-contiguous, one operand = two 16-byte loads at
// row*stride + kbase and row*stride + kbase + 16, kbase = (lane>>4)*8.
__device__ __forceinline__ int kmap(int lane, int e) {
  return ((lane >> 4) << 3) + (e & 7) + ((e >> 3) << 4);
}
__device__ __forceinline__ v16bf load_op_g(const unsigned short* base16B) {
  B32x8 u;
  u.lo = *(const u32x4*)(base16B);
  u.hi = *(const u32x4*)(base16B + 16);
  return __builtin_bit_cast(v16bf, u);
}
// 32-bit C/D layout: elem r -> row M = r + 8*(lane>>4), col N = lane&15

// ---------------------------------------------------------------------
// Pack kernels: one-time fp32 -> bf16 conversion / transposition so the
// hot loops consume WMMA operands with plain b128 loads.
// ---------------------------------------------------------------------

// K fp32 [n,b,v,hd] -> bf16 same layout (elementwise, 4 per thread)
__global__ void __launch_bounds__(256)
pack_k_kernel(const float* __restrict__ K, unsigned short* __restrict__ Kb) {
  const size_t i = ((size_t)blockIdx.x * 256 + threadIdx.x) * 4;
  f32x4 v = *(const f32x4*)(K + i);
  unsigned int w0 = (unsigned)f2bfbits(v.x) | ((unsigned)f2bfbits(v.y) << 16);
  unsigned int w1 = (unsigned)f2bfbits(v.z) | ((unsigned)f2bfbits(v.w) << 16);
  u32x2 w = { w0, w1 };
  *(u32x2*)(Kb + i) = w;
}

// V fp32 [n,b,v,hd] -> bf16 transposed Vt[n,b,hd,v] via 32x32 LDS tile
__global__ void __launch_bounds__(256)
pack_v_kernel(const float* __restrict__ V, unsigned short* __restrict__ Vt) {
  __shared__ unsigned short tile[32][33];
  const int t  = threadIdx.x;
  const int v0 = blockIdx.x * 32;
  const int b  = blockIdx.y;
  const int n  = blockIdx.z;
  const float* src = V + (((size_t)n * BATCH + b) * VNUM + v0) * HEAD_DIM;
  // phase 1: coalesced read, convert, LDS
  {
    const int i = t * 4;                 // i = v*32 + d
    const int v = i >> 5, d = i & 31;
    f32x4 x = *(const f32x4*)(src + (size_t)v * HEAD_DIM + d);
    tile[v][d + 0] = f2bfbits(x.x);
    tile[v][d + 1] = f2bfbits(x.y);
    tile[v][d + 2] = f2bfbits(x.z);
    tile[v][d + 3] = f2bfbits(x.w);
  }
  __syncthreads();
  // phase 2: transposed, packed store
  {
    const int j = t * 4;                 // j = d*32 + v
    const int d = j >> 5, v = j & 31;
    unsigned int w0 = (unsigned)tile[v + 0][d] | ((unsigned)tile[v + 1][d] << 16);
    unsigned int w1 = (unsigned)tile[v + 2][d] | ((unsigned)tile[v + 3][d] << 16);
    u32x2 w = { w0, w1 };
    unsigned short* dst = Vt + (((size_t)n * BATCH + b) * HEAD_DIM + d) * VNUM + v0 + v;
    *(u32x2*)dst = w;
  }
}

// nn_O fp32 [k,n] -> bf16 transposed Ot[n,k]
__global__ void __launch_bounds__(256)
pack_w_kernel(const float* __restrict__ nnO, unsigned short* __restrict__ Ot) {
  const int idx = blockIdx.x * 256 + threadIdx.x;  // 65536 total
  const int k = idx >> 8, n = idx & 255;
  Ot[(size_t)n * HIDDEN + k] = f2bfbits(nnO[(size_t)k * HIDDEN + n]);
}

// ---------------------------------------------------------------------
// Kernel: per-head Q projection.  Qp[n,b,q,:] = q[b,q,:] @ nn_Q[n], bf16 out.
// One wave computes a 16x32 output tile.  (Run once; inline converts OK.)
// ---------------------------------------------------------------------
__global__ void __launch_bounds__(32)
qproj_kernel(const float* __restrict__ q,
             const float* __restrict__ nnQ,
             unsigned short* __restrict__ Qp) {
  const int lane = threadIdx.x;
  const int q0 = blockIdx.x * 16;
  const int b  = blockIdx.y;
  const int n  = blockIdx.z;
  const int m  = lane & 15;

  const float* A  = q   + ((size_t)b * QNUM + q0) * HIDDEN;
  const float* Bw = nnQ + (size_t)n * HIDDEN * HEAD_DIM;

  v8f acc0 = {}; v8f acc1 = {};
  for (int k0 = 0; k0 < HIDDEN; k0 += 32) {
    v16bf a, b0, b1;
#pragma unroll
    for (int e = 0; e < 16; ++e) {
      const int k = k0 + kmap(lane, e);
      a[e]  = f2bf(A[m * HIDDEN + k]);
      b0[e] = f2bf(Bw[k * HEAD_DIM + m]);
      b1[e] = f2bf(Bw[k * HEAD_DIM + 16 + m]);
    }
    acc0 = __builtin_amdgcn_wmma_f32_16x16x32_bf16(false, a, false, b0,
                                                   (short)0, acc0, false, false);
    acc1 = __builtin_amdgcn_wmma_f32_16x16x32_bf16(false, a, false, b1,
                                                   (short)0, acc1, false, false);
  }
  unsigned short* O = Qp + (((size_t)n * BATCH + b) * QNUM + q0) * HEAD_DIM;
#pragma unroll
  for (int r = 0; r < 8; ++r) {
    const int mr = r + ((lane >> 4) << 3);
    O[mr * HEAD_DIM + m]      = f2bfbits(acc0[r]);
    O[mr * HEAD_DIM + 16 + m] = f2bfbits(acc1[r]);
  }
}

// ---------------------------------------------------------------------
// Flash attention.  One wave owns a 32-row Q tile for (head, batch):
// each loaded K/V operand feeds TWO score / output WMMAs (halves L2
// traffic per row vs a 16-row tile).  Per 32 v-indices:
//   4 b128 K loads -> 4 WMMA scores (2 row-tiles x 2 col-chunks)
//   masked online softmax, P staged via LDS (32x32 bf16)
//   4 b128 V loads -> 4 WMMA outputs
// ---------------------------------------------------------------------
__global__ void __launch_bounds__(32)
attn_kernel(const unsigned short* __restrict__ Qp,
            const unsigned short* __restrict__ Kb,
            const unsigned short* __restrict__ Vt,
            const unsigned char* __restrict__ mask,
            unsigned short* __restrict__ xb) {
  __shared__ __align__(16) unsigned short Pt[32 * 32];  // P tile, bf16

  const int lane = threadIdx.x;
  const int q0 = blockIdx.x * 32;
  const int b  = blockIdx.y;
  const int n  = blockIdx.z;
  const int m  = lane & 15;
  const int kbase = (lane >> 4) << 3;
  const float alpha = 0.17677669529663687f;  // 1/sqrt(32)
  const float NEG_INF = -__builtin_inff();

  // Q tiles (A operands): rows q0..q0+15 and q0+16..q0+31
  const unsigned short* Qt = Qp + (((size_t)n * BATCH + b) * QNUM + q0) * HEAD_DIM;
  v16bf aq[2];
#pragma unroll
  for (int t = 0; t < 2; ++t)
    aq[t] = load_op_g(Qt + (size_t)(t * 16 + m) * HEAD_DIM + kbase);

  const unsigned short* Kp = Kb + ((size_t)n * BATCH + b) * VNUM * HEAD_DIM;
  const unsigned short* Vp = Vt + ((size_t)n * BATCH + b) * HEAD_DIM * VNUM;
  const unsigned char*  Mb = mask + ((size_t)b * QNUM + q0) * VNUM;

  v8f o[2][2] = {};                 // [row-tile][col-chunk]
  float rmax[2][8], rsum[2][8];
#pragma unroll
  for (int t = 0; t < 2; ++t)
#pragma unroll
    for (int r = 0; r < 8; ++r) { rmax[t][r] = NEG_INF; rsum[t][r] = 0.0f; }

  for (int v0 = 0; v0 < VNUM; v0 += 32) {
    // prefetch next block (global_prefetch_b8, speculative)
    __builtin_prefetch(Kp + (size_t)(v0 + 32 + m) * HEAD_DIM + kbase, 0, 0);
    __builtin_prefetch(Vp + (size_t)m * VNUM + v0 + 32 + kbase, 0, 0);

    // ---- scores: S(32x32) = Q * K^T, K operands reused for both row-tiles ----
    const v16bf bk0 = load_op_g(Kp + (size_t)(v0 + m) * HEAD_DIM + kbase);
    const v16bf bk1 = load_op_g(Kp + (size_t)(v0 + 16 + m) * HEAD_DIM + kbase);
    v8f zero = {};
    v8f s[2][2];
#pragma unroll
    for (int t = 0; t < 2; ++t) {
      s[t][0] = __builtin_amdgcn_wmma_f32_16x16x32_bf16(false, aq[t], false, bk0,
                                                        (short)0, zero, false, false);
      s[t][1] = __builtin_amdgcn_wmma_f32_16x16x32_bf16(false, aq[t], false, bk1,
                                                        (short)0, zero, false, false);
    }

    // ---- mask + scale + online softmax (C layout rows) ----
#pragma unroll
    for (int t = 0; t < 2; ++t) {
#pragma unroll
      for (int r = 0; r < 8; ++r) {
        const int row = t * 16 + r + kbase;           // q-row within 32-tile
        const size_t mrow = (size_t)row * VNUM + v0;
        float e0 = Mb[mrow + m]      ? NEG_INF : s[t][0][r] * alpha;
        float e1 = Mb[mrow + 16 + m] ? NEG_INF : s[t][1][r] * alpha;

        float tmax = fmaxf(e0, e1);
#pragma unroll
        for (int off = 1; off < 16; off <<= 1)
          tmax = fmaxf(tmax, __shfl_xor(tmax, off, 32));   // 16-lane half
        const float nmax  = fmaxf(rmax[t][r], tmax);
        const float scale = __expf(rmax[t][r] - nmax);
        const float p0 = __expf(e0 - nmax);
        const float p1 = __expf(e1 - nmax);
        float psum = p0 + p1;
#pragma unroll
        for (int off = 1; off < 16; off <<= 1)
          psum += __shfl_xor(psum, off, 32);
        rsum[t][r] = rsum[t][r] * scale + psum;
        rmax[t][r] = nmax;
        o[t][0][r] *= scale;
        o[t][1][r] *= scale;
        Pt[row * 32 + m]      = f2bfbits(p0);
        Pt[row * 32 + 16 + m] = f2bfbits(p1);
      }
    }
    __syncthreads();

    // ---- O += P(32x32) * V(32x32), V operands reused for both row-tiles ----
    const v16bf bv0 = load_op_g(Vp + (size_t)m        * VNUM + v0 + kbase);
    const v16bf bv1 = load_op_g(Vp + (size_t)(16 + m) * VNUM + v0 + kbase);
#pragma unroll
    for (int t = 0; t < 2; ++t) {
      B32x8 up;
      up.lo = *(const u32x4*)(&Pt[(t * 16 + m) * 32 + kbase]);
      up.hi = *(const u32x4*)(&Pt[(t * 16 + m) * 32 + kbase + 16]);
      const v16bf ap = __builtin_bit_cast(v16bf, up);
      o[t][0] = __builtin_amdgcn_wmma_f32_16x16x32_bf16(false, ap, false, bv0,
                                                        (short)0, o[t][0], false, false);
      o[t][1] = __builtin_amdgcn_wmma_f32_16x16x32_bf16(false, ap, false, bv1,
                                                        (short)0, o[t][1], false, false);
    }
    __syncthreads();
  }

  // ---- normalize, write x[b,q, n*32+d] in bf16 for the O-projection ----
  unsigned short* X = xb + ((size_t)b * QNUM + q0) * HIDDEN + n * HEAD_DIM;
#pragma unroll
  for (int t = 0; t < 2; ++t) {
#pragma unroll
    for (int r = 0; r < 8; ++r) {
      const int row = t * 16 + r + kbase;
      const float inv = 1.0f / rsum[t][r];
      X[row * HIDDEN + m]      = f2bfbits(o[t][0][r] * inv);
      X[row * HIDDEN + 16 + m] = f2bfbits(o[t][1][r] * inv);
    }
  }
}

// ---------------------------------------------------------------------
// Output projection: out[4096,256] = x @ nn_O.  A and B already bf16 with
// K-contiguous rows -> pure b128 operand loads.
// ---------------------------------------------------------------------
__global__ void __launch_bounds__(32)
oproj_kernel(const unsigned short* __restrict__ xb,
             const unsigned short* __restrict__ Ot,
             float* __restrict__ out) {
  const int lane = threadIdx.x;
  const int row0 = blockIdx.x * 16;
  const int col0 = blockIdx.y * 16;
  const int m = lane & 15;
  const int kbase = (lane >> 4) << 3;

  v8f acc = {};
  for (int k0 = 0; k0 < HIDDEN; k0 += 32) {
    const v16bf a  = load_op_g(xb + (size_t)(row0 + m) * HIDDEN + k0 + kbase);
    const v16bf bw = load_op_g(Ot + (size_t)(col0 + m) * HIDDEN + k0 + kbase);
    acc = __builtin_amdgcn_wmma_f32_16x16x32_bf16(false, a, false, bw,
                                                  (short)0, acc, false, false);
  }
#pragma unroll
  for (int r = 0; r < 8; ++r) {
    const int mr = r + kbase;
    out[(size_t)(row0 + mr) * HIDDEN + col0 + m] = acc[r];
  }
}

// ---------------------------------------------------------------------
extern "C" void kernel_launch(void* const* d_in, const int* in_sizes, int n_in,
                              void* d_out, int out_size, void* d_ws, size_t ws_size,
                              hipStream_t stream) {
  (void)in_sizes; (void)n_in; (void)out_size; (void)ws_size;
  const float*         q    = (const float*)d_in[0];
  const float*         K    = (const float*)d_in[1];
  const float*         V    = (const float*)d_in[2];
  const unsigned char* mask = (const unsigned char*)d_in[3];  // jnp.bool_ = 1 byte
  const float*         nnQ  = (const float*)d_in[4];
  const float*         nnO  = (const float*)d_in[5];
  float* out = (float*)d_out;

  // workspace: all bf16 (unsigned short) buffers, 16B-aligned sections
  const size_t KV_ELEMS = (size_t)HEADS * BATCH * VNUM * HEAD_DIM;   // 8M
  const size_t QP_ELEMS = (size_t)HEADS * BATCH * QNUM * HEAD_DIM;   // 1M
  const size_t XB_ELEMS = (size_t)BATCH * QNUM * HIDDEN;             // 1M
  unsigned short* Kb = (unsigned short*)d_ws;
  unsigned short* Vt = Kb + KV_ELEMS;
  unsigned short* Qp = Vt + KV_ELEMS;
  unsigned short* xb = Qp + QP_ELEMS;
  unsigned short* Ot = xb + XB_ELEMS;

  // one-time packing passes
  pack_k_kernel<<<dim3((unsigned)(KV_ELEMS / 1024)), 256, 0, stream>>>(K, Kb);
  pack_v_kernel<<<dim3(VNUM / 32, BATCH, HEADS), 256, 0, stream>>>(V, Vt);
  pack_w_kernel<<<dim3((HIDDEN * HIDDEN) / 256), 256, 0, stream>>>(nnO, Ot);

  dim3 gq(QNUM / 16, BATCH, HEADS);
  qproj_kernel<<<gq, 32, 0, stream>>>(q, nnQ, Qp);

  dim3 ga(QNUM / 32, BATCH, HEADS);
  attn_kernel<<<ga, 32, 0, stream>>>(Qp, Kb, Vt, mask, xb);

  dim3 go((BATCH * QNUM) / 16, HIDDEN / 16);
  oproj_kernel<<<go, 32, 0, stream>>>(xb, Ot, out);
}
